// ReasoningManager_37864431681832
// MI455X (gfx1250) — matrix operand order repaired and checked by reading, not compiled, not run on hardware
//
#include <hip/hip_runtime.h>
#include <math.h>
#include <stdint.h>

#define B_ 4
#define S_ 512
#define H_ 1024
#define KF 3
#define MAXSIMS 32

// ---------------- WMMA types (gfx1250, wave32) ----------------
typedef __attribute__((ext_vector_type(16))) __bf16 v16bf_t;
typedef __attribute__((ext_vector_type(8)))  float  v8f_t;

__device__ __forceinline__ float gelu_exact(float x) {
    return 0.5f * x * (1.0f + erff(x * 0.7071067811865475f));
}

__device__ __forceinline__ float gumbel_noise(unsigned step, unsigned b, unsigned s) {
    unsigned long long z = ((unsigned long long)(step * 0x9E3779B1u + 0x85EBCA6Bu) << 32)
                         ^ ((unsigned long long)b * 0xC2B2AE3Dull)
                         ^ ((unsigned long long)s * 0x27D4EB2Full)
                         ^ 0x9E3779B97F4A7C15ull;
    z = (z ^ (z >> 30)) * 0xBF58476D1CE4E5B9ull;
    z = (z ^ (z >> 27)) * 0x94D049BB133111EBull;
    z ^= (z >> 31);
    float u = ((float)(unsigned)(z >> 40) + 0.5f) * (1.0f / 16777216.0f);
    return -logf(-logf(u));
}

// split x0,x1 into bf16 hi/lo pairs packed as dwords (2 x bf16 each)
__device__ __forceinline__ void split_pack(float x0, float x1, unsigned& hp, unsigned& lp) {
    __bf16 h0 = (__bf16)x0, h1 = (__bf16)x1;
    __bf16 l0 = (__bf16)(x0 - (float)h0), l1 = (__bf16)(x1 - (float)h1);
    hp = (unsigned)__builtin_bit_cast(unsigned short, h0)
       | ((unsigned)__builtin_bit_cast(unsigned short, h1) << 16);
    lp = (unsigned)__builtin_bit_cast(unsigned short, l0)
       | ((unsigned)__builtin_bit_cast(unsigned short, l1) << 16);
}

// =================================================================
// Big GEMM: C[M,N] = act(A[M,K] @ W[K,N] + bias) (+ residual)
// fp32 in/out; internally bf16 hi/lo split with 3 WMMAs per tile
// (hi*hi + hi*lo + lo*hi) -> ~fp32 accuracy at bf16 matrix rate.
// LDS tiles are stored FRAGMENT-MAJOR ([tile][lane][elem]) so each
// lane fetches its whole 16-elem bf16 fragment with 2x ds_load_b128.
// Split-K A: rows from A0 for k<Ksplit, A1 otherwise.
// Block = 256 threads (8 waves). Tile 128(M) x 64(N) x 32(K).
// Wave owns 2 m-tiles x 2 n-tiles = 4 accumulators, 12 WMMA/k-block.
// Requires M%128==0, N%64==0, K%32==0, Ksplit%32==0.
// =================================================================
#define BM 128
#define BN 64
#define BK 32
#define NMT 8   // 16-row tiles per block
#define NNT 4   // 16-col tiles per block

#define WMMA_BF16(A, Bv, Cv) \
    __builtin_amdgcn_wmma_f32_16x16x32_bf16(false, (A), false, (Bv), (short)0, (Cv), false, false)

__global__ __launch_bounds__(256)
void gemm_bf16x3(const float* __restrict__ A0, const float* __restrict__ A1, int Ksplit,
                 int lda0, int lda1,
                 const float* __restrict__ W, int ldb,
                 const float* __restrict__ bias,
                 const float* __restrict__ residual,
                 float* __restrict__ C, int ldc,
                 int M, int N, int K, int doGelu)
{
    // fragment-major staging: [tile][lane][16 bf16] (32B per lane, 128b-load friendly)
    __shared__ v16bf_t aHi[NMT][32];
    __shared__ v16bf_t aLo[NMT][32];
    __shared__ v16bf_t bHi[NNT][32];
    __shared__ v16bf_t bLo[NNT][32];

    const int tid  = threadIdx.x;
    const int wave = tid >> 5;
    const int lane = tid & 31;
    const int half = lane >> 4;
    const int l16  = lane & 15;

    const int m0 = blockIdx.x * BM;
    const int n0 = blockIdx.y * BN;

    const int mtA = (wave >> 1);      // 0..3
    const int mtB = mtA + 4;          // 4..7
    const int nt0 = (wave & 1) * 2;   // 0 or 2
    const int nt1 = nt0 + 1;

    v8f_t acc00 = {0.f,0.f,0.f,0.f,0.f,0.f,0.f,0.f};
    v8f_t acc01 = acc00, acc10 = acc00, acc11 = acc00;

    for (int k0 = 0; k0 < K; k0 += BK) {
        // ---- A tile 128x32: thread handles 8 k-pairs (float2 loads) ----
        #pragma unroll
        for (int it = 0; it < 8; ++it) {
            int r = (tid >> 4) + it * 16;      // row in tile 0..127
            int k = (tid & 15) * 2;            // even k in 0..30
            int gk = k0 + k;
            int gr = m0 + r;
            const float* src = (gk < Ksplit) ? (A0 + (size_t)gr * lda0 + gk)
                                             : (A1 + (size_t)gr * lda1 + (gk - Ksplit));
            float2 x = *(const float2*)src;
            unsigned hp, lp;
            split_pack(x.x, x.y, hp, lp);
            // ISA A-fragment mapping (16-bit A 16x32):
            //   lane = (k>>3 & 1)*16 + (r&15);  elem = (k&7) + ((k&16)>>1)
            int mt = r >> 4;
            int ln = ((k >> 3) & 1) * 16 + (r & 15);
            int e  = (k & 7) + ((k & 16) >> 1);
            ((unsigned*)&aHi[mt][ln])[e >> 1] = hp;
            ((unsigned*)&aLo[mt][ln])[e >> 1] = lp;
        }
        // ---- B tile 32x64: thread handles 4 kr-pairs (coalesced in n) ----
        #pragma unroll
        for (int it = 0; it < 4; ++it) {
            int c  = tid & 63;                    // col in tile
            int kr = ((tid >> 6) + it * 4) * 2;   // even k-row in 0..30
            float x0 = W[(size_t)(k0 + kr)     * ldb + (n0 + c)];
            float x1 = W[(size_t)(k0 + kr + 1) * ldb + (n0 + c)];
            unsigned hp, lp;
            split_pack(x0, x1, hp, lp);
            // ISA B-fragment mapping (16-bit B 32x16):
            //   lane = (kr>>3 & 1)*16 + (c&15);  elem = (kr&7) + ((kr&16)>>1)
            int nt = c >> 4;
            int ln = ((kr >> 3) & 1) * 16 + (c & 15);
            int e  = (kr & 7) + ((kr & 16) >> 1);
            ((unsigned*)&bHi[nt][ln])[e >> 1] = hp;
            ((unsigned*)&bLo[nt][ln])[e >> 1] = lp;
        }
        __syncthreads();

        // ---- wide fragment loads: 2x ds_load_b128 each ----
        v16bf_t a0h = aHi[mtA][lane], a0l = aLo[mtA][lane];
        v16bf_t a1h = aHi[mtB][lane], a1l = aLo[mtB][lane];
        v16bf_t b0h = bHi[nt0][lane], b0l = bLo[nt0][lane];
        v16bf_t b1h = bHi[nt1][lane], b1l = bLo[nt1][lane];

        // 12 WMMAs, 4 independent accumulator chains (hides RAW hazard)
        acc00 = WMMA_BF16(a0h, b0h, acc00);
        acc01 = WMMA_BF16(a0h, b1h, acc01);
        acc10 = WMMA_BF16(a1h, b0h, acc10);
        acc11 = WMMA_BF16(a1h, b1h, acc11);
        acc00 = WMMA_BF16(a0h, b0l, acc00);
        acc01 = WMMA_BF16(a0h, b1l, acc01);
        acc10 = WMMA_BF16(a1h, b0l, acc10);
        acc11 = WMMA_BF16(a1h, b1l, acc11);
        acc00 = WMMA_BF16(a0l, b0h, acc00);
        acc01 = WMMA_BF16(a0l, b1h, acc01);
        acc10 = WMMA_BF16(a1l, b0h, acc10);
        acc11 = WMMA_BF16(a1l, b1h, acc11);

        __syncthreads();
    }

    // ---- epilogue: C/D 16x16 layout: VGPR r -> row 8*half+r, col = lane%16 ----
    #pragma unroll
    for (int q = 0; q < 4; ++q) {
        const int mt  = (q < 2) ? mtA : mtB;
        const int nt  = (q & 1) ? nt1 : nt0;
        v8f_t acc = (q == 0) ? acc00 : (q == 1) ? acc01 : (q == 2) ? acc10 : acc11;
        #pragma unroll
        for (int r = 0; r < 8; ++r) {
            int row = m0 + mt * 16 + half * 8 + r;
            int col = n0 + nt * 16 + l16;
            float v = acc[r] + bias[col];
            if (doGelu) v = gelu_exact(v);
            if (residual) v += residual[(size_t)row * ldc + col];
            C[(size_t)row * ldc + col] = v;
        }
    }
}

// =================================================================
// Small helper kernels (VALU; latency-bound sequential work)
// =================================================================
__global__ void k_zero(float* p, int n) {
    int i = blockIdx.x * 256 + threadIdx.x;
    if (i < n) p[i] = 0.f;
}

__global__ void k_init_cfac(float* cfac) {
    if (threadIdx.x < B_) cfac[threadIdx.x] = 1.0f;
}

__global__ void k_copy(float* dst, const float* src, int n) {
    int i = blockIdx.x * 256 + threadIdx.x;
    if (i < n) dst[i] = src[i];
}

__global__ void k_rootmean(const float* __restrict__ root, float* __restrict__ rm) {
    int i = blockIdx.x * 256 + threadIdx.x;
    if (i >= B_ * H_) return;
    int b = i / H_, h = i % H_;
    float acc = 0.f;
    const float* p = root + (size_t)b * S_ * H_ + h;
    for (int s = 0; s < S_; ++s) acc += p[(size_t)s * H_];
    rm[i] = acc * (1.0f / S_);
}

// wbar[k] = mean_j Wp2[k,j]; bbar = mean(bp2)   (5 columns)
__global__ void k_wbar(const float* __restrict__ Wp2, const float* __restrict__ bp2,
                       float* __restrict__ wbar, float* __restrict__ bbar) {
    int i = blockIdx.x * 256 + threadIdx.x;
    if (i < H_) {
        float a = 0.f;
        for (int j = 0; j < 5; ++j) a += Wp2[i * 5 + j];
        wbar[i] = a * 0.2f;
    }
    if (i == 0) {
        float a = 0.f;
        for (int j = 0; j < 5; ++j) a += bp2[j];
        bbar[0] = a * 0.2f;
    }
}

// Y[b,n] = act( sum_{k<K0} X0[b*s0+k]*W[k,n] + sum_{K0<=k<Kt} X1[b*s1+k-K0]*W[k,n] + bias[n] )
__global__ void k_mm2(const float* __restrict__ X0, int s0,
                      const float* __restrict__ X1, int s1,
                      int K0, int Kt,
                      const float* __restrict__ W, const float* __restrict__ bias,
                      float* __restrict__ Y, int doGelu) {
    int i = blockIdx.x * 256 + threadIdx.x;
    if (i >= B_ * H_) return;
    int b = i / H_, n = i % H_;
    float acc = bias[n];
    const float* x0 = X0 + (size_t)b * s0;
    for (int k = 0; k < K0; ++k) acc += x0[k] * W[(size_t)k * H_ + n];
    const float* x1 = X1 + (size_t)b * s1;
    for (int k = K0; k < Kt; ++k) acc += x1[k - K0] * W[(size_t)k * H_ + n];
    Y[i] = doGelu ? gelu_exact(acc) : acc;
}

// classifier: logits_j = sum_k hc[b,k]*Wc2[k,j] + bc2[j]; simCounts[b]=SIM_OPTIONS[argmax]
__global__ void k_cls_argmax(const float* __restrict__ hc, const float* __restrict__ Wc2,
                             const float* __restrict__ bc2, int* __restrict__ simCounts) {
    __shared__ float red[256];
    const int simopt[5] = {10, 10, 16, 24, 32};
    int b = blockIdx.x, t = threadIdx.x;
    float best = -3.0e38f; int bi = 0;
    for (int j = 0; j < 5; ++j) {
        float p = 0.f;
        for (int k = t; k < H_; k += 256) p += hc[b * H_ + k] * Wc2[k * 5 + j];
        red[t] = p; __syncthreads();
        for (int off = 128; off > 0; off >>= 1) {
            if (t < off) red[t] += red[t + off];
            __syncthreads();
        }
        if (t == 0) {
            float lg = red[0] + bc2[j];
            if (lg > best) { best = lg; bi = j; }
        }
        __syncthreads();
    }
    if (t == 0) simCounts[b] = simopt[bi];
}

// focus[row] = dot(P1[row,:], wbar) + bbar   (one wave per row)
__global__ void k_focus(const float* __restrict__ P1, const float* __restrict__ wbar,
                        const float* __restrict__ bbar, float* __restrict__ focus) {
    int wave = threadIdx.x >> 5, lane = threadIdx.x & 31;
    int row = blockIdx.x * 8 + wave;
    if (row >= B_ * S_) return;
    float p = 0.f;
    const float* r = P1 + (size_t)row * H_;
    for (int k = lane; k < H_; k += 32) p += r[k] * wbar[k];
    for (int off = 16; off > 0; off >>= 1) p += __shfl_down(p, off, 32);
    if (lane == 0) focus[row] = p + bbar[0];
}

__global__ void k_logsoftmax(const float* __restrict__ focus, const int* __restrict__ amask,
                             float* __restrict__ logp) {
    __shared__ float red[512];
    int b = blockIdx.x, s = threadIdx.x;
    float x = focus[b * S_ + s];
    if (amask[b * S_ + s] == 0) x = -1.0e9f;
    red[s] = x; __syncthreads();
    for (int off = 256; off > 0; off >>= 1) {
        if (s < off) red[s] = fmaxf(red[s], red[s + off]);
        __syncthreads();
    }
    float mx = red[0]; __syncthreads();
    red[s] = expf(x - mx); __syncthreads();
    for (int off = 256; off > 0; off >>= 1) {
        if (s < off) red[s] += red[s + off];
        __syncthreads();
    }
    float lse = mx + logf(red[0]);
    logp[b * S_ + s] = x - lse;
}

// top-3 of (logp + gumbel) per batch; 3 sequential argmax passes
__global__ void k_topk(const float* __restrict__ logp, int step, int* __restrict__ pos) {
    __shared__ float sv[512];
    __shared__ int   si[512];
    int b = blockIdx.x, s = threadIdx.x;
    float sc = logp[b * S_ + s] + gumbel_noise((unsigned)step, (unsigned)b, (unsigned)s);
    for (int k = 0; k < KF; ++k) {
        sv[s] = sc; si[s] = s; __syncthreads();
        for (int off = 256; off > 0; off >>= 1) {
            if (s < off) {
                if (sv[s + off] > sv[s] || (sv[s + off] == sv[s] && si[s + off] < si[s])) {
                    sv[s] = sv[s + off]; si[s] = si[s + off];
                }
            }
            __syncthreads();
        }
        int best = si[0];
        if (s == 0) pos[b * KF + k] = best;
        if (s == best) sc = -3.0e38f;
        __syncthreads();
    }
}

// tf[b,h] = root[b, pos[b,kidx], h]
__global__ void k_gather_focus(const float* __restrict__ root, const int* __restrict__ pos,
                               int kidx, float* __restrict__ tf) {
    int i = blockIdx.x * 256 + threadIdx.x;
    if (i >= B_ * H_) return;
    int b = i / H_, h = i % H_;
    int p = pos[b * KF + kidx];
    tf[i] = root[((size_t)b * S_ + p) * H_ + h];
}

// new = hstep @ Wt2 + bt2; store newrow; curMean += (new - root[pos])/S
__global__ void k_stepg2(const float* __restrict__ hstep, const float* __restrict__ Wt2,
                         const float* __restrict__ bt2, const float* __restrict__ root,
                         const int* __restrict__ pos, int kidx,
                         float* __restrict__ newrow, float* __restrict__ curMean) {
    int i = blockIdx.x * 256 + threadIdx.x;
    if (i >= B_ * H_) return;
    int b = i / H_, n = i % H_;
    float acc = bt2[n];
    const float* h = hstep + (size_t)b * H_;
    for (int k = 0; k < H_; ++k) acc += h[k] * Wt2[(size_t)k * H_ + n];
    newrow[((size_t)b * KF + kidx) * H_ + n] = acc;
    int p = pos[b * KF + kidx];
    float rr = root[((size_t)b * S_ + p) * H_ + n];
    curMean[i] += (acc - rr) * (1.0f / S_);
}

// val = dot(aBuf[b,:], Wa2) + ba2; aw[b] = (step<simCounts)? sigmoid(val):0; cfac += aw
__global__ void k_val2(const float* __restrict__ aBuf, const float* __restrict__ Wa2,
                       const float* __restrict__ ba2, const int* __restrict__ simCounts,
                       int step, float* __restrict__ aw, float* __restrict__ cfac) {
    __shared__ float red[256];
    int b = blockIdx.x, t = threadIdx.x;
    float p = 0.f;
    for (int n = t; n < H_; n += 256) p += aBuf[b * H_ + n] * Wa2[n];
    red[t] = p; __syncthreads();
    for (int off = 128; off > 0; off >>= 1) {
        if (t < off) red[t] += red[t + off];
        __syncthreads();
    }
    if (t == 0) {
        float v = red[0] + ba2[0];
        float w = 1.0f / (1.0f + expf(-v));
        float a = (step < simCounts[b]) ? w : 0.f;
        aw[b] = a;
        cfac[b] += a;
    }
}

// D[b, pos_k, :] += aw[b] * (newrow - root[b, pos_k, :])
__global__ void k_accD(const float* __restrict__ newrow, const float* __restrict__ root,
                       const int* __restrict__ pos, const float* __restrict__ aw,
                       float* __restrict__ D) {
    int i = blockIdx.x * 256 + threadIdx.x;
    if (i >= B_ * KF * H_) return;
    int b = i / (KF * H_);
    int rem = i % (KF * H_);
    int kidx = rem / H_, n = rem % H_;
    int p = pos[b * KF + kidx];
    size_t idx = ((size_t)b * S_ + p) * H_ + n;
    D[idx] += aw[b] * (newrow[((size_t)b * KF + kidx) * H_ + n] - root[idx]);
}

// D <- accumulated = cfac[b]*root + D   (in place)
__global__ void k_accfinal(const float* __restrict__ root, const float* __restrict__ cfac,
                           float* __restrict__ D) {
    int i = blockIdx.x * 256 + threadIdx.x;
    if (i >= B_ * S_ * H_) return;
    int b = i / (S_ * H_);
    D[i] = cfac[b] * root[i] + D[i];
}

// =================================================================
extern "C" void kernel_launch(void* const* d_in, const int* in_sizes, int n_in,
                              void* d_out, int out_size, void* d_ws, size_t ws_size,
                              hipStream_t stream) {
    const float* root  = (const float*)d_in[0];
    const int*   amask = (const int*)  d_in[1];
    const float* Wc1 = (const float*)d_in[2];
    const float* bc1 = (const float*)d_in[3];
    const float* Wc2 = (const float*)d_in[4];
    const float* bc2 = (const float*)d_in[5];
    const float* Wp1 = (const float*)d_in[6];
    const float* bp1 = (const float*)d_in[7];
    const float* Wp2 = (const float*)d_in[8];
    const float* bp2 = (const float*)d_in[9];
    const float* Wt1 = (const float*)d_in[10];
    const float* bt1 = (const float*)d_in[11];
    const float* Wt2 = (const float*)d_in[12];
    const float* bt2 = (const float*)d_in[13];
    const float* Wa1 = (const float*)d_in[14];
    const float* ba1 = (const float*)d_in[15];
    const float* Wa2 = (const float*)d_in[16];
    const float* ba2 = (const float*)d_in[17];
    const float* Wg1 = (const float*)d_in[18];
    const float* bg1 = (const float*)d_in[19];
    const float* Wg2 = (const float*)d_in[20];
    const float* bg2 = (const float*)d_in[21];
    float* out = (float*)d_out;

    // ---- workspace layout (floats) ----
    float* ws = (float*)d_ws;
    const size_t BSH = (size_t)B_ * S_ * H_;       // 2M floats
    float* P1       = ws;                          // [BS,H] policy hidden / reused as G1
    float* D        = P1 + BSH;                    // [B,S,H] sparse delta / accumulated
    float* sm       = D + BSH;
    float* rootMean = sm; sm += B_ * H_;
    float* curMean  = sm; sm += B_ * H_;
    float* wbar     = sm; sm += H_;
    float* bbar     = sm; sm += 4;
    float* hc       = sm; sm += B_ * H_;           // classifier hidden
    float* focus    = sm; sm += B_ * S_;
    float* logp     = sm; sm += B_ * S_;
    float* tf       = sm; sm += B_ * H_;           // focus row gather
    float* hstep    = sm; sm += B_ * H_;           // step hidden
    float* aBuf     = sm; sm += B_ * H_;           // value hidden
    float* newrow   = sm; sm += B_ * KF * H_;
    float* aw       = sm; sm += 4;
    float* cfac     = sm; sm += 4;
    int*   simCounts = (int*)sm;
    int*   pos       = simCounts + 8;

    const int TB = 256;
    int gBH  = (B_ * H_ + TB - 1) / TB;
    int gBKH = (B_ * KF * H_ + TB - 1) / TB;
    int gBSH = (int)((BSH + TB - 1) / TB);

    // ---- init ----
    k_zero<<<gBSH, TB, 0, stream>>>(D, (int)BSH);
    k_init_cfac<<<1, 32, 0, stream>>>(cfac);
    k_rootmean<<<gBH, TB, 0, stream>>>(root, rootMean);
    k_wbar<<<(H_ + TB - 1) / TB, TB, 0, stream>>>(Wp2, bp2, wbar, bbar);

    // ---- classifier head -> simCounts ----
    k_mm2<<<gBH, TB, 0, stream>>>(root, S_ * H_, root, S_ * H_, H_, H_, Wc1, bc1, hc, 1);
    k_cls_argmax<<<B_, TB, 0, stream>>>(hc, Wc2, bc2, simCounts);

    // ---- policy GEMM1 (WMMA): P1 = gelu(root @ Wp1 + bp1), [2048x1024x1024] ----
    dim3 gg((B_ * S_) / BM, H_ / BN);
    gemm_bf16x3<<<gg, TB, 0, stream>>>(root, root, H_, H_, H_, Wp1, H_, bp1, nullptr,
                                       P1, H_, B_ * S_, H_, H_, 1);
    // policy GEMM2 folded into GEMV via column-mean of Wp2 (exact)
    k_focus<<<(B_ * S_) / 8, TB, 0, stream>>>(P1, wbar, bbar, focus);
    k_logsoftmax<<<B_, 512, 0, stream>>>(focus, amask, logp);

    // ---- 32 sequential simulation steps ----
    for (int s = 0; s < MAXSIMS; ++s) {
        k_topk<<<B_, 512, 0, stream>>>(logp, s, pos);
        k_copy<<<gBH, TB, 0, stream>>>(curMean, rootMean, B_ * H_);
        for (int k = 0; k < KF; ++k) {
            k_gather_focus<<<gBH, TB, 0, stream>>>(root, pos, k, tf);
            // h = gelu( concat(curMean, root[pos]) @ Wt1 + bt1 )
            k_mm2<<<gBH, TB, 0, stream>>>(curMean, H_, tf, H_, H_, 2 * H_, Wt1, bt1, hstep, 1);
            // new = h @ Wt2 + bt2 ; incremental mean update
            k_stepg2<<<gBH, TB, 0, stream>>>(hstep, Wt2, bt2, root, pos, k, newrow, curMean);
        }
        // val path: a = gelu( concat(rootMean, curMean) @ Wa1 + ba1 )
        k_mm2<<<gBH, TB, 0, stream>>>(rootMean, H_, curMean, H_, H_, 2 * H_, Wa1, ba1, aBuf, 1);
        k_val2<<<B_, TB, 0, stream>>>(aBuf, Wa2, ba2, simCounts, s, aw, cfac);
        k_accD<<<gBKH, TB, 0, stream>>>(newrow, root, pos, aw, D);
    }

    // ---- accumulated = cfac[b]*root + D (in place into D) ----
    k_accfinal<<<gBSH, TB, 0, stream>>>(root, cfac, D);

    // ---- final GEMMs (WMMA). combined = [root | accumulated] via split-K A ----
    // G1 = gelu(combined @ Wg1 + bg1): M=2048, N=1024, K=2048  (reuse P1)
    gemm_bf16x3<<<gg, TB, 0, stream>>>(root, D, H_, H_, H_, Wg1, H_, bg1, nullptr,
                                       P1, H_, B_ * S_, H_, 2 * H_, 1);
    // out = root + (G1 @ Wg2 + bg2)
    gemm_bf16x3<<<gg, TB, 0, stream>>>(P1, P1, H_, H_, H_, Wg2, H_, bg2, root,
                                       out, H_, B_ * S_, H_, H_, 0);
}
